// MrT5Block_57157424775173
// MI455X (gfx1250) — compile-verified, matrix operand order checked by reading, not run on hardware
//
#include <hip/hip_runtime.h>
#include <hip/hip_bf16.h>

// ---------------- types ----------------
typedef __bf16 bf16;
typedef bf16  bf16x8  __attribute__((ext_vector_type(8)));
typedef bf16  bf16x16 __attribute__((ext_vector_type(16)));
typedef float floatx8 __attribute__((ext_vector_type(8)));

#define BB 2
#define SS 4096
#define DD 512
#define HH 8
#define DKV 64
#define DFF 2048
#define MM (BB*SS)   // 8192

__device__ __forceinline__ bf16 f2bf(float x) {
    unsigned u = __builtin_bit_cast(unsigned, x);
    unsigned r = u + 0x7FFFu + ((u >> 16) & 1u);
    unsigned short hs = (unsigned short)(r >> 16);
    return __builtin_bit_cast(bf16, hs);
}
__device__ __forceinline__ unsigned short f2bfbits(float x) {
    unsigned u = __builtin_bit_cast(unsigned, x);
    unsigned r = u + 0x7FFFu + ((u >> 16) & 1u);
    return (unsigned short)(r >> 16);
}
__device__ __forceinline__ bf16x16 load16(const bf16* p0, const bf16* p1) {
    bf16x8 a = *(const bf16x8*)p0;
    bf16x8 b = *(const bf16x8*)p1;
    bf16x16 r;
#pragma unroll
    for (int i = 0; i < 8; ++i) { r[i] = a[i]; r[i + 8] = b[i]; }
    return r;
}
__device__ __forceinline__ floatx8 wmma_bf16(bf16x16 a, bf16x16 b, floatx8 c) {
    return __builtin_amdgcn_wmma_f32_16x16x32_bf16(false, a, false, b, (short)0, c,
                                                   false, false);
}
__device__ __forceinline__ floatx8 zero8() {
    floatx8 z = {0.f, 0.f, 0.f, 0.f, 0.f, 0.f, 0.f, 0.f};
    return z;
}

// CDNA5 async global->LDS copy (16B per lane), tracked by ASYNCcnt.
__device__ __forceinline__ void async_copy_b128(const bf16* g, bf16* l) {
    unsigned lofs = (unsigned)(size_t)l;  // low 32 bits of generic shared ptr = DS addr
    asm volatile("global_load_async_to_lds_b128 %0, %1, off"
                 :: "v"(lofs), "v"(g) : "memory");
}
__device__ __forceinline__ void wait_async0() {
    asm volatile("s_wait_asynccnt 0x0" ::: "memory");
}

// ---------------- small kernels ----------------
// fp32 [K,N] -> bf16 transposed [N,K]  (one-time; makes all WMMA B-operands K-contiguous)
__global__ void cvtT_kernel(const float* __restrict__ in, bf16* __restrict__ out,
                            int K, int N) {
    int i = blockIdx.x * 256 + threadIdx.x;
    if (i >= K * N) return;
    int k = i / N, n = i % N;
    out[(size_t)n * K + k] = f2bf(in[i]);
}

// T5 bidirectional bucket bias, Toeplitz table: tab[h*8192 + (rel+4095)]
__global__ void bias_table_kernel(const float* __restrict__ rel_bias, float* __restrict__ tab) {
    int idx = blockIdx.x * 256 + threadIdx.x;
    if (idx >= HH * 8191) return;
    int h = idx / 8191, r = idx % 8191;
    int rel = r - 4095;
    int ret = rel > 0 ? 16 : 0;         // num_buckets/2 = 16
    int n = rel < 0 ? -rel : rel;
    int bucket;
    if (n < 8) {                        // max_exact = 8
        bucket = ret + n;
    } else {
        int v = 8 + (int)(__logf((float)n * 0.125f) * (8.0f / __logf(16.0f)));
        if (v > 15) v = 15;
        bucket = ret + v;
    }
    tab[h * 8192 + r] = rel_bias[bucket * HH + h];
}

// one wave per row of 512 columns; fp32 in, bf16*(rms weight) out
__global__ void __launch_bounds__(256) rmsnorm_kernel(const float* __restrict__ x,
                                                      const float* __restrict__ w,
                                                      bf16* __restrict__ out, int rows) {
    int wave = threadIdx.x >> 5, lane = threadIdx.x & 31;
    int row = blockIdx.x * 8 + wave;
    if (row >= rows) return;
    const float* xr = x + (size_t)row * DD;
    float vals[16];
    float ss = 0.f;
#pragma unroll
    for (int i = 0; i < 16; ++i) {
        float v = xr[lane + i * 32];
        vals[i] = v;
        ss += v * v;
    }
#pragma unroll
    for (int m = 16; m >= 1; m >>= 1) ss += __shfl_xor(ss, m, 32);
    float rs = rsqrtf(ss * (1.f / (float)DD) + 1e-6f);
#pragma unroll
    for (int i = 0; i < 16; ++i) {
        int c = lane + i * 32;
        out[(size_t)row * DD + c] = f2bf(vals[i] * rs * w[c]);
    }
}

// ---------------- WMMA GEMM (async double-buffered LDS) ----------------
// C[M,N] = A[M,K] @ W[K,N] with W pre-transposed as Wt[N,K].
// Block 256 thr = 8 waves (4x2); block tile 128x128; wave tile 32x64; K-step 32.
// MODE 0: bf16 row-major    MODE 1: per-head [B,H,S,64]   MODE 2: V^T [B,H,64,S]
// MODE 3: f32 + residual    MODE 4: relu -> bf16
template <int MODE>
__global__ void __launch_bounds__(256) gemm_kernel(const bf16* __restrict__ A,
                                                   const bf16* __restrict__ Wt,
                                                   const float* __restrict__ resid,
                                                   void* __restrict__ outp,
                                                   int M, int N, int K) {
    __shared__ __align__(16) bf16 As[2][128 * 40];  // [row][k], pad 40
    __shared__ __align__(16) bf16 Bs[2][128 * 40];  // [n][k],  pad 40
    const int tid = threadIdx.x, lane = tid & 31, wave = tid >> 5;
    const int wm = wave >> 1, wn = wave & 1;
    const int n0 = blockIdx.x * 128, m0 = blockIdx.y * 128;
    const int l15 = lane & 15, kb8 = (lane >> 4) * 8;

    floatx8 acc[2][4];
#pragma unroll
    for (int qt = 0; qt < 2; ++qt)
#pragma unroll
        for (int nt = 0; nt < 4; ++nt) acc[qt][nt] = zero8();

    const int pr = tid >> 1, pk = (tid & 1) * 16;  // staging assignment
    const int nIter = K >> 5;

    // prefetch tile 0
    {
        const bf16* ga = A + (size_t)(m0 + pr) * K + pk;
        const bf16* gb = Wt + (size_t)(n0 + pr) * K + pk;
        async_copy_b128(ga,     &As[0][pr * 40 + pk]);
        async_copy_b128(ga + 8, &As[0][pr * 40 + pk + 8]);
        async_copy_b128(gb,     &Bs[0][pr * 40 + pk]);
        async_copy_b128(gb + 8, &Bs[0][pr * 40 + pk + 8]);
    }

    for (int it = 0; it < nIter; ++it) {
        const int bb = it & 1;
        wait_async0();
        __syncthreads();
        if (it + 1 < nIter) {  // overlap next-tile async copy with WMMA
            int kk = (it + 1) << 5;
            const bf16* ga = A + (size_t)(m0 + pr) * K + kk + pk;
            const bf16* gb = Wt + (size_t)(n0 + pr) * K + kk + pk;
            async_copy_b128(ga,     &As[bb ^ 1][pr * 40 + pk]);
            async_copy_b128(ga + 8, &As[bb ^ 1][pr * 40 + pk + 8]);
            async_copy_b128(gb,     &Bs[bb ^ 1][pr * 40 + pk]);
            async_copy_b128(gb + 8, &Bs[bb ^ 1][pr * 40 + pk + 8]);
        }
        bf16x16 af[2];
#pragma unroll
        for (int qt = 0; qt < 2; ++qt) {
            int row = wm * 32 + qt * 16 + l15;
            af[qt] = load16(&As[bb][row * 40 + kb8], &As[bb][row * 40 + kb8 + 16]);
        }
#pragma unroll
        for (int nt = 0; nt < 4; ++nt) {
            int col = wn * 64 + nt * 16 + l15;
            bf16x16 bfr = load16(&Bs[bb][col * 40 + kb8], &Bs[bb][col * 40 + kb8 + 16]);
#pragma unroll
            for (int qt = 0; qt < 2; ++qt)
                acc[qt][nt] = wmma_bf16(af[qt], bfr, acc[qt][nt]);
        }
    }

#pragma unroll
    for (int qt = 0; qt < 2; ++qt) {
#pragma unroll
        for (int nt = 0; nt < 4; ++nt) {
#pragma unroll
            for (int i = 0; i < 8; ++i) {
                int m = m0 + wm * 32 + qt * 16 + i + 8 * (lane >> 4);
                int n = n0 + wn * 64 + nt * 16 + l15;
                float v = acc[qt][nt][i];
                if constexpr (MODE == 0) {
                    ((bf16*)outp)[(size_t)m * N + n] = f2bf(v);
                } else if constexpr (MODE == 1) {
                    int b = m >> 12, s = m & 4095, h = n >> 6, d = n & 63;
                    ((bf16*)outp)[(((size_t)(b * HH + h) * SS) + s) * DKV + d] = f2bf(v);
                } else if constexpr (MODE == 2) {
                    int b = m >> 12, s = m & 4095, h = n >> 6, d = n & 63;
                    ((bf16*)outp)[(((size_t)(b * HH + h) * DKV) + d) * SS + s] = f2bf(v);
                } else if constexpr (MODE == 3) {
                    ((float*)outp)[(size_t)m * N + n] = resid[(size_t)m * N + n] + v;
                } else {  // MODE 4
                    ((bf16*)outp)[(size_t)m * N + n] = f2bf(v > 0.f ? v : 0.f);
                }
            }
        }
    }
}

// ---------------- Flash attention with softmax1 ----------------
// Block = 128 thr = 4 waves; block covers (b, h, 128 q rows); wave covers 32 q rows.
// K/V fragments are streamed (loaded in pairs and consumed immediately) to stay
// under the 256-VGPR direct window (avoids s_set_vgpr_msb in the hot loop).
__global__ void __launch_bounds__(128) attn_kernel(const bf16* __restrict__ Qh,
                                                   const bf16* __restrict__ Kh,
                                                   const bf16* __restrict__ Vt,
                                                   const float* __restrict__ btab,
                                                   const float* __restrict__ gate,
                                                   bf16* __restrict__ ctx) {
    __shared__ float bias_s[192];
    __shared__ __align__(16) unsigned short Pl[4][32 * 40];  // per-wave P relayout
    const int tid = threadIdx.x, lane = tid & 31, wave = tid >> 5;
    const int qblk = blockIdx.x * 128;
    const int h = blockIdx.y, b = blockIdx.z;
    const int q0 = qblk + wave * 32;
    const size_t head = (size_t)(b * HH + h) * SS * DKV;
    const bf16* Qp = Qh + head;
    const bf16* Kp = Kh + head;
    const bf16* Vp = Vt + head;  // [64][4096]
    const float* btabh = btab + h * 8192;
    const float* gateb = gate + (size_t)b * SS;

    const int l15 = lane & 15;
    const int kb8 = (lane >> 4) * 8;

    // Q fragments: 2 q-strips x 2 K-halves (live across the whole loop)
    bf16x16 qf[2][2];
#pragma unroll
    for (int qt = 0; qt < 2; ++qt) {
        const bf16* qr = Qp + (size_t)(q0 + qt * 16 + l15) * DKV;
#pragma unroll
        for (int kh = 0; kh < 2; ++kh)
            qf[qt][kh] = load16(qr + kh * 32 + kb8, qr + kh * 32 + kb8 + 16);
    }

    floatx8 Oa[2][4];
#pragma unroll
    for (int qt = 0; qt < 2; ++qt)
#pragma unroll
        for (int nt = 0; nt < 4; ++nt) Oa[qt][nt] = zero8();
    float rm[2][8], rl[2][8];
#pragma unroll
    for (int qt = 0; qt < 2; ++qt)
#pragma unroll
        for (int i = 0; i < 8; ++i) { rm[qt][i] = -1e30f; rl[qt][i] = 0.f; }

    for (int kb = 0; kb < SS; kb += 32) {
        __syncthreads();
        {   // stage Toeplitz bias slice (rel = key - q); 160 entries needed
            int base = kb - qblk - 127 + 4095;
            int gi = base + tid;
            gi = gi < 0 ? 0 : (gi > 8190 ? 8190 : gi);
            bias_s[tid] = btabh[gi];
            if (tid < 32) {
                int gi2 = base + 128 + tid;
                gi2 = gi2 < 0 ? 0 : (gi2 > 8190 ? 8190 : gi2);
                bias_s[128 + tid] = btabh[gi2];
            }
        }
        __syncthreads();

        // ---- scores: stream K fragments per key-tile, 8 WMMAs ----
        floatx8 sacc[2][2];
#pragma unroll
        for (int nt = 0; nt < 2; ++nt) {
            const bf16* kr = Kp + (size_t)(kb + nt * 16 + l15) * DKV;
            bf16x16 kf0 = load16(kr + kb8, kr + kb8 + 16);
            bf16x16 kf1 = load16(kr + 32 + kb8, kr + 32 + kb8 + 16);
#pragma unroll
            for (int qt = 0; qt < 2; ++qt) {
                floatx8 t = wmma_bf16(qf[qt][0], kf0, zero8());
                sacc[qt][nt] = wmma_bf16(qf[qt][1], kf1, t);
            }
        }

        // ---- + position bias + delete-gate ----
        float gv0 = gateb[kb + l15];
        float gv1 = gateb[kb + 16 + l15];
#pragma unroll
        for (int qt = 0; qt < 2; ++qt)
#pragma unroll
            for (int nt = 0; nt < 2; ++nt) {
                float gv = nt ? gv1 : gv0;
#pragma unroll
                for (int i = 0; i < 8; ++i) {
                    int offq = wave * 32 + qt * 16 + i + 8 * (lane >> 4);
                    int offk = nt * 16 + l15;
                    sacc[qt][nt][i] += bias_s[offk + 127 - offq] + gv;
                }
            }

        // ---- online softmax1 update ----
#pragma unroll
        for (int qt = 0; qt < 2; ++qt) {
#pragma unroll
            for (int i = 0; i < 8; ++i) {
                float tm = fmaxf(sacc[qt][0][i], sacc[qt][1][i]);
                tm = fmaxf(tm, __shfl_xor(tm, 1, 32));
                tm = fmaxf(tm, __shfl_xor(tm, 2, 32));
                tm = fmaxf(tm, __shfl_xor(tm, 4, 32));
                tm = fmaxf(tm, __shfl_xor(tm, 8, 32));
                float mnew = fmaxf(rm[qt][i], tm);
                float alpha = __expf(rm[qt][i] - mnew);
                rm[qt][i] = mnew;
                float p0 = __expf(sacc[qt][0][i] - mnew);
                float p1 = __expf(sacc[qt][1][i] - mnew);
                sacc[qt][0][i] = p0;
                sacc[qt][1][i] = p1;
                float ts = p0 + p1;
                ts += __shfl_xor(ts, 1, 32);
                ts += __shfl_xor(ts, 2, 32);
                ts += __shfl_xor(ts, 4, 32);
                ts += __shfl_xor(ts, 8, 32);
                rl[qt][i] = rl[qt][i] * alpha + ts;
#pragma unroll
                for (int nt = 0; nt < 4; ++nt) Oa[qt][nt][i] *= alpha;
            }
        }

        // ---- relayout P (C-layout -> A-layout) via per-wave LDS ----
#pragma unroll
        for (int qt = 0; qt < 2; ++qt)
#pragma unroll
            for (int nt = 0; nt < 2; ++nt)
#pragma unroll
                for (int i = 0; i < 8; ++i) {
                    int r = qt * 16 + i + 8 * (lane >> 4);
                    Pl[wave][r * 40 + nt * 16 + l15] = f2bfbits(sacc[qt][nt][i]);
                }
        bf16x16 pf[2];
#pragma unroll
        for (int qt = 0; qt < 2; ++qt) {
            int r = qt * 16 + l15;
            pf[qt] = load16((const bf16*)&Pl[wave][r * 40 + kb8],
                            (const bf16*)&Pl[wave][r * 40 + kb8 + 16]);
        }

        // ---- O += P V : stream V fragments in pairs, 8 WMMAs ----
#pragma unroll
        for (int np = 0; np < 2; ++np) {
            const bf16* vr0 = Vp + (size_t)((2 * np) * 16 + l15) * SS + kb;
            const bf16* vr1 = Vp + (size_t)((2 * np + 1) * 16 + l15) * SS + kb;
            bf16x16 vfa = load16(vr0 + kb8, vr0 + kb8 + 16);
            bf16x16 vfb = load16(vr1 + kb8, vr1 + kb8 + 16);
#pragma unroll
            for (int qt = 0; qt < 2; ++qt) {
                Oa[qt][2 * np]     = wmma_bf16(pf[qt], vfa, Oa[qt][2 * np]);
                Oa[qt][2 * np + 1] = wmma_bf16(pf[qt], vfb, Oa[qt][2 * np + 1]);
            }
        }
    }

    // ---- softmax1 denominator: l + exp(-m) ----
#pragma unroll
    for (int qt = 0; qt < 2; ++qt)
#pragma unroll
        for (int i = 0; i < 8; ++i) {
            float inv = 1.f / (rl[qt][i] + __expf(-rm[qt][i]));
#pragma unroll
            for (int nt = 0; nt < 4; ++nt) Oa[qt][nt][i] *= inv;
        }
    // ---- write ctx [B,S,H*DKV] ----
#pragma unroll
    for (int qt = 0; qt < 2; ++qt)
#pragma unroll
        for (int nt = 0; nt < 4; ++nt)
#pragma unroll
            for (int i = 0; i < 8; ++i) {
                int q = q0 + qt * 16 + i + 8 * (lane >> 4);
                int n = h * DKV + nt * 16 + l15;
                ctx[((size_t)b * SS + q) * DD + n] = f2bf(Oa[qt][nt][i]);
            }
}

// ---------------- launcher ----------------
extern "C" void kernel_launch(void* const* d_in, const int* in_sizes, int n_in,
                              void* d_out, int out_size, void* d_ws, size_t ws_size,
                              hipStream_t stream) {
    (void)in_sizes; (void)n_in; (void)out_size; (void)ws_size;
    const float* hidden = (const float*)d_in[0];
    const float* gate   = (const float*)d_in[1];
    const float* ln1    = (const float*)d_in[2];
    const float* qw     = (const float*)d_in[3];
    const float* kw     = (const float*)d_in[4];
    const float* vw     = (const float*)d_in[5];
    const float* ow     = (const float*)d_in[6];
    const float* relb   = (const float*)d_in[7];
    const float* ln2    = (const float*)d_in[8];
    const float* wi     = (const float*)d_in[9];
    const float* wo     = (const float*)d_in[10];

    char* ws = (char*)d_ws;
    size_t off = 0;
    auto take = [&](size_t bytes) -> char* {
        char* p = ws + off;
        off = (off + bytes + 255) & ~(size_t)255;
        return p;
    };
    bf16*  qwb  = (bf16*)take((size_t)DD * DD * 2);    // all weights transposed [N,K]
    bf16*  kwb  = (bf16*)take((size_t)DD * DD * 2);
    bf16*  vwb  = (bf16*)take((size_t)DD * DD * 2);
    bf16*  owb  = (bf16*)take((size_t)DD * DD * 2);
    bf16*  wib  = (bf16*)take((size_t)DD * DFF * 2);
    bf16*  wob  = (bf16*)take((size_t)DFF * DD * 2);
    float* btab = (float*)take((size_t)HH * 8192 * 4);
    bf16*  xb   = (bf16*)take((size_t)MM * DD * 2);
    bf16*  Qhd  = (bf16*)take((size_t)MM * DD * 2);
    bf16*  Khd  = (bf16*)take((size_t)MM * DD * 2);
    bf16*  Vtd  = (bf16*)take((size_t)MM * DD * 2);
    bf16*  ctxb = (bf16*)take((size_t)MM * DD * 2);
    float* hbuf = (float*)take((size_t)MM * DD * 4);
    bf16*  yb   = (bf16*)take((size_t)MM * DD * 2);
    bf16*  ffb  = (bf16*)take((size_t)MM * DFF * 2);

    // transposed bf16 weight conversions (one-time)
    cvtT_kernel<<<(DD * DD + 255) / 256, 256, 0, stream>>>(qw, qwb, DD, DD);
    cvtT_kernel<<<(DD * DD + 255) / 256, 256, 0, stream>>>(kw, kwb, DD, DD);
    cvtT_kernel<<<(DD * DD + 255) / 256, 256, 0, stream>>>(vw, vwb, DD, DD);
    cvtT_kernel<<<(DD * DD + 255) / 256, 256, 0, stream>>>(ow, owb, DD, DD);
    cvtT_kernel<<<(DD * DFF + 255) / 256, 256, 0, stream>>>(wi, wib, DD, DFF);
    cvtT_kernel<<<(DFF * DD + 255) / 256, 256, 0, stream>>>(wo, wob, DFF, DD);
    bias_table_kernel<<<(HH * 8191 + 255) / 256, 256, 0, stream>>>(relb, btab);

    // ln1 -> xb (bf16)
    rmsnorm_kernel<<<MM / 8, 256, 0, stream>>>(hidden, ln1, xb, MM);

    // QKV projections (head layouts; V transposed)
    dim3 gQKV(DD / 128, MM / 128);
    gemm_kernel<1><<<gQKV, 256, 0, stream>>>(xb, qwb, nullptr, Qhd, MM, DD, DD);
    gemm_kernel<1><<<gQKV, 256, 0, stream>>>(xb, kwb, nullptr, Khd, MM, DD, DD);
    gemm_kernel<2><<<gQKV, 256, 0, stream>>>(xb, vwb, nullptr, Vtd, MM, DD, DD);

    // flash attention with softmax1
    attn_kernel<<<dim3(SS / 128, HH, BB), 128, 0, stream>>>(Qhd, Khd, Vtd, btab, gate, ctxb);

    // h = hidden + ctx @ o_w
    gemm_kernel<3><<<dim3(DD / 128, MM / 128), 256, 0, stream>>>(ctxb, owb, hidden, hbuf,
                                                                 MM, DD, DD);
    // ln2 -> yb
    rmsnorm_kernel<<<MM / 8, 256, 0, stream>>>(hbuf, ln2, yb, MM);

    // FFN: relu(y @ wi) then out = h + (.) @ wo
    gemm_kernel<4><<<dim3(DFF / 128, MM / 128), 256, 0, stream>>>(yb, wib, nullptr, ffb,
                                                                  MM, DFF, DD);
    gemm_kernel<3><<<dim3(DD / 128, MM / 128), 256, 0, stream>>>(ffb, wob, hbuf, d_out,
                                                                 MM, DD, DFF);
}
